// DynamicRoutingFusion_18176301596855
// MI455X (gfx1250) — compile-verified, matrix operand order checked by reading, not run on hardware
//
#include <hip/hip_runtime.h>

// ---------------- CDNA5 WMMA types ----------------
typedef __attribute__((ext_vector_type(16))) __bf16 v16bf;
typedef __attribute__((ext_vector_type(8)))  float  v8f;
typedef __attribute__((ext_vector_type(4)))  unsigned int u32x4;

union V16 { v16bf bf; u32x4 q[2]; };

#define B_  8
#define L_  2048
#define LV_ 256
#define H_  1024
#define R_  (B_ * L_)   // 16384 rows

__device__ __forceinline__ unsigned short bf16_trunc(float x) {
    return (unsigned short)(__float_as_uint(x) >> 16);
}
__device__ __forceinline__ float bf16_to_f(unsigned short h) {
    return __uint_as_float(((unsigned int)h) << 16);
}

// Per-lane tile load for 16-bit 16x32 operand (A layout; W loaded identically
// since B = W^T has the symmetric layout). elems[0..7] = [row, kb..kb+7],
// elems[8..15] = [row, kb+16..kb+23]; kb already includes the +8 upper-half shift.
__device__ __forceinline__ v16bf load_tile(const unsigned short* __restrict__ p,
                                           int row, int kb, int ld) {
    V16 t;
    const unsigned short* base = p + (size_t)row * ld + kb;
    t.q[0] = *reinterpret_cast<const u32x4*>(base);
    t.q[1] = *reinterpret_cast<const u32x4*>(base + 16);
    return t.bf;
}

__device__ __forceinline__ v8f wmma_bf16(v16bf a, v16bf b, v8f c) {
    return __builtin_amdgcn_wmma_f32_16x16x32_bf16(
        /*neg_a=*/false, a, /*neg_b=*/false, b,
        /*c_mod=*/(short)0, c, /*reuse_a=*/false, /*reuse_b=*/false);
}

// ---------------- small prep kernels ----------------

// vision_pooled[b,h] = mean_l vision_feat[b,l,h]
__global__ void pool_kernel(const float* __restrict__ vf, float* __restrict__ vpool) {
    int i = blockIdx.x * blockDim.x + threadIdx.x;      // B*H threads
    int b = i >> 10, h = i & (H_ - 1);
    const float* p = vf + (size_t)b * LV_ * H_ + h;
    float s = 0.f;
    for (int l = 0; l < LV_; ++l) s += p[(size_t)l * H_];
    vpool[i] = s * (1.0f / (float)LV_);
}

// vision_proj[b,o] = vpool[b,:] . Wv[o,:] + bv[o]
__global__ void vproj_kernel(const float* __restrict__ vpool, const float* __restrict__ Wv,
                             const float* __restrict__ bv, float* __restrict__ vproj) {
    int i = blockIdx.x * blockDim.x + threadIdx.x;
    int b = i >> 10, o = i & (H_ - 1);
    const float* x = vpool + (b << 10);
    const float* w = Wv + (size_t)o * H_;
    float s = 0.f;
    for (int h = 0; h < H_; ++h) s += x[h] * w[h];
    vproj[i] = s + bv[o];
}

// vpg[b,o] = vision_proj[b,:] . Wg[o, H: ]   (bg added later in the gate epilogue)
__global__ void vpg_kernel(const float* __restrict__ vproj, const float* __restrict__ Wg,
                           float* __restrict__ vpg) {
    int i = blockIdx.x * blockDim.x + threadIdx.x;
    int b = i >> 10, o = i & (H_ - 1);
    const float* x = vproj + (b << 10);
    const float* w = Wg + (size_t)o * (2 * H_) + H_;
    float s = 0.f;
    for (int h = 0; h < H_; ++h) s += x[h] * w[h];
    vpg[i] = s;
}

// fp32 -> bf16 hi/lo split (contiguous)
__global__ void cvt_split(const float* __restrict__ src, unsigned short* __restrict__ hi,
                          unsigned short* __restrict__ lo, size_t n) {
    size_t i = (size_t)blockIdx.x * blockDim.x + threadIdx.x;
    if (i >= n) return;
    float x = src[i];
    unsigned short h = bf16_trunc(x);
    hi[i] = h;
    lo[i] = bf16_trunc(x - bf16_to_f(h));
}

// strided variant (for Wg[:, :H] with row stride 2H); dst is dense rows x cols
__global__ void cvt_split_strided(const float* __restrict__ src, int ldsrc,
                                  unsigned short* __restrict__ hi, unsigned short* __restrict__ lo,
                                  int rows, int cols) {
    int i = blockIdx.x * blockDim.x + threadIdx.x;
    if (i >= rows * cols) return;
    int r = i / cols, c = i - r * cols;
    float x = src[(size_t)r * ldsrc + c];
    unsigned short h = bf16_trunc(x);
    hi[i] = h;
    lo[i] = bf16_trunc(x - bf16_to_f(h));
}

// ---------------- main WMMA GEMM: C = A(RxK) * W(NxK)^T ----------------
// A, W provided as bf16 hi/lo split; fp32 accumulate via 3-product correction.
// mode 0: out = acc + bt  -> stored as bf16 hi/lo pair (text_proj for GEMM2)
// mode 1: gate_pre = acc + rw*vpg + bg; fused = text + sigmoid(gate_pre)*rw*vproj -> outp
__global__ __launch_bounds__(256)
void gemm_kernel(const unsigned short* __restrict__ Ahi, const unsigned short* __restrict__ Alo,
                 const unsigned short* __restrict__ Whi, const unsigned short* __restrict__ Wlo,
                 int mode, const float* __restrict__ bias,
                 unsigned short* __restrict__ TPhi, unsigned short* __restrict__ TPlo,
                 const float* __restrict__ rw, const float* __restrict__ vpg,
                 const float* __restrict__ vproj, const float* __restrict__ text,
                 float* __restrict__ outp)
{
    const int K = H_;
    const int lane  = threadIdx.x & 31;
    const int wave  = threadIdx.x >> 5;
    const int l15   = lane & 15;
    const int khalf = (lane >> 4) << 3;            // 0 or 8
    const int n0    = blockIdx.x * 64;             // 4 N-subtiles of 16
    const int r0    = (blockIdx.y * 8 + wave) * 32;// 2 M-subtiles of 16

    v8f acc[2][4];
#pragma unroll
    for (int t = 0; t < 2; ++t)
#pragma unroll
        for (int j = 0; j < 4; ++j) acc[t][j] = (v8f){0.f,0.f,0.f,0.f,0.f,0.f,0.f,0.f};

    for (int k0 = 0; k0 < K; k0 += 32) {
        const int kb = k0 + khalf;
        v16bf ah0 = load_tile(Ahi, r0 + l15,      kb, K);
        v16bf al0 = load_tile(Alo, r0 + l15,      kb, K);
        v16bf ah1 = load_tile(Ahi, r0 + 16 + l15, kb, K);
        v16bf al1 = load_tile(Alo, r0 + 16 + l15, kb, K);
#pragma unroll
        for (int j = 0; j < 4; ++j) {
            v16bf wh = load_tile(Whi, n0 + j * 16 + l15, kb, K);
            v16bf wl = load_tile(Wlo, n0 + j * 16 + l15, kb, K);
            // A*W ~= Ah*Wh + Ah*Wl + Al*Wh  (fp32 accumulate)
            acc[0][j] = wmma_bf16(ah0, wh, acc[0][j]);
            acc[0][j] = wmma_bf16(ah0, wl, acc[0][j]);
            acc[0][j] = wmma_bf16(al0, wh, acc[0][j]);
            acc[1][j] = wmma_bf16(ah1, wh, acc[1][j]);
            acc[1][j] = wmma_bf16(ah1, wl, acc[1][j]);
            acc[1][j] = wmma_bf16(al1, wh, acc[1][j]);
        }
    }

    // C layout: lane covers n = l15; VGPR v covers m = v + 8*(lane>=16)
    const int mbase = (lane >> 4) << 3;
#pragma unroll
    for (int t = 0; t < 2; ++t) {
#pragma unroll
        for (int j = 0; j < 4; ++j) {
            const int n = n0 + j * 16 + l15;
#pragma unroll
            for (int v = 0; v < 8; ++v) {
                const int row = r0 + t * 16 + mbase + v;
                float val = acc[t][j][v];
                if (mode == 0) {
                    val += bias[n];
                    unsigned short h = bf16_trunc(val);
                    TPhi[(size_t)row * H_ + n] = h;
                    TPlo[(size_t)row * H_ + n] = bf16_trunc(val - bf16_to_f(h));
                } else {
                    const int b = row >> 11;          // row / L
                    float r  = rw[row];
                    float gp = val + r * vpg[(b << 10) + n] + bias[n];
                    float gate = 1.f / (1.f + __expf(-gp));
                    float wv = r * vproj[(b << 10) + n];
                    outp[(size_t)row * H_ + n] = text[(size_t)row * H_ + n] + gate * wv;
                }
            }
        }
    }
}

// agreement[row] = text_proj[row,:] . vision_proj[b,:]   (tp reconstructed hi+lo)
__global__ void agree_kernel(const unsigned short* __restrict__ tph,
                             const unsigned short* __restrict__ tpl,
                             const float* __restrict__ vproj, float* __restrict__ agree) {
    int gid  = blockIdx.x * blockDim.x + threadIdx.x;
    int row  = gid >> 5;
    int lane = gid & 31;
    int b    = row >> 11;
    const float* vp = vproj + (b << 10);
    float s = 0.f;
    for (int o = lane; o < H_; o += 32) {
        float tp = bf16_to_f(tph[(size_t)row * H_ + o]) + bf16_to_f(tpl[(size_t)row * H_ + o]);
        s += tp * vp[o];
    }
#pragma unroll
    for (int off = 16; off; off >>= 1) s += __shfl_xor(s, off, 32);
    if (lane == 0) agree[row] = s;
}

// rw[b,:] = softmax(2*agreement[b,:]) over L   (final-iteration coupling = 2*agreement)
__global__ void softmax_kernel(const float* __restrict__ agree, float* __restrict__ rw) {
    __shared__ float sd[256];
    const int b = blockIdx.x, tid = threadIdx.x;
    const float* a = agree + b * L_;
    float m = -3.0e38f;
    for (int l = tid; l < L_; l += 256) m = fmaxf(m, 2.f * a[l]);
    sd[tid] = m; __syncthreads();
    for (int s = 128; s; s >>= 1) { if (tid < s) sd[tid] = fmaxf(sd[tid], sd[tid + s]); __syncthreads(); }
    m = sd[0]; __syncthreads();
    float sum = 0.f;
    for (int l = tid; l < L_; l += 256) sum += __expf(2.f * a[l] - m);
    sd[tid] = sum; __syncthreads();
    for (int s = 128; s; s >>= 1) { if (tid < s) sd[tid] += sd[tid + s]; __syncthreads(); }
    float inv = 1.f / sd[0];
    for (int l = tid; l < L_; l += 256) rw[b * L_ + l] = __expf(2.f * a[l] - m) * inv;
}

// in-place row LayerNorm on d_out (one block per row, 4 elems/thread)
__global__ void ln_kernel(float* __restrict__ out, const float* __restrict__ gamma,
                          const float* __restrict__ beta) {
    __shared__ float sd[256];
    const int row = blockIdx.x, tid = threadIdx.x;
    float* p = out + (size_t)row * H_;
    float x[4];
#pragma unroll
    for (int i = 0; i < 4; ++i) x[i] = p[tid + i * 256];
    float s = x[0] + x[1] + x[2] + x[3];
    sd[tid] = s; __syncthreads();
    for (int k = 128; k; k >>= 1) { if (tid < k) sd[tid] += sd[tid + k]; __syncthreads(); }
    const float mu = sd[0] * (1.0f / (float)H_); __syncthreads();
    float vs = 0.f;
#pragma unroll
    for (int i = 0; i < 4; ++i) { float d = x[i] - mu; vs += d * d; }
    sd[tid] = vs; __syncthreads();
    for (int k = 128; k; k >>= 1) { if (tid < k) sd[tid] += sd[tid + k]; __syncthreads(); }
    const float inv = rsqrtf(sd[0] * (1.0f / (float)H_) + 1e-5f);
#pragma unroll
    for (int i = 0; i < 4; ++i) {
        int c = tid + i * 256;
        p[c] = (x[i] - mu) * inv * gamma[c] + beta[c];
    }
}

// ---------------- host ----------------
extern "C" void kernel_launch(void* const* d_in, const int* in_sizes, int n_in,
                              void* d_out, int out_size, void* d_ws, size_t ws_size,
                              hipStream_t stream) {
    const float* text  = (const float*)d_in[0];
    const float* vf    = (const float*)d_in[1];
    const float* Wt    = (const float*)d_in[2];
    const float* bt    = (const float*)d_in[3];
    const float* Wv    = (const float*)d_in[4];
    const float* bv    = (const float*)d_in[5];
    const float* Wg    = (const float*)d_in[6];
    const float* bg    = (const float*)d_in[7];
    const float* gamma = (const float*)d_in[8];
    const float* beta  = (const float*)d_in[9];
    float* out = (float*)d_out;

    char* ws = (char*)d_ws;
    size_t off = 0;
    auto alloc = [&](size_t bytes) -> void* {
        void* p = ws + off;
        off += (bytes + 255) & ~(size_t)255;
        return p;
    };
    unsigned short* Ahi  = (unsigned short*)alloc((size_t)R_ * H_ * 2);
    unsigned short* Alo  = (unsigned short*)alloc((size_t)R_ * H_ * 2);
    unsigned short* TPhi = (unsigned short*)alloc((size_t)R_ * H_ * 2);
    unsigned short* TPlo = (unsigned short*)alloc((size_t)R_ * H_ * 2);
    unsigned short* Wthi = (unsigned short*)alloc((size_t)H_ * H_ * 2);
    unsigned short* Wtlo = (unsigned short*)alloc((size_t)H_ * H_ * 2);
    unsigned short* Wghi = (unsigned short*)alloc((size_t)H_ * H_ * 2);
    unsigned short* Wglo = (unsigned short*)alloc((size_t)H_ * H_ * 2);
    float* vpool = (float*)alloc((size_t)B_ * H_ * 4);
    float* vproj = (float*)alloc((size_t)B_ * H_ * 4);
    float* vpg   = (float*)alloc((size_t)B_ * H_ * 4);
    float* agree = (float*)alloc((size_t)R_ * 4);
    float* rw    = (float*)alloc((size_t)R_ * 4);

    pool_kernel <<<(B_ * H_) / 256, 256, 0, stream>>>(vf, vpool);
    vproj_kernel<<<(B_ * H_) / 256, 256, 0, stream>>>(vpool, Wv, bv, vproj);
    vpg_kernel  <<<(B_ * H_) / 256, 256, 0, stream>>>(vproj, Wg, vpg);

    cvt_split<<<((size_t)R_ * H_) / 256, 256, 0, stream>>>(text, Ahi, Alo, (size_t)R_ * H_);
    cvt_split_strided<<<(H_ * H_) / 256, 256, 0, stream>>>(Wt, H_,     Wthi, Wtlo, H_, H_);
    cvt_split_strided<<<(H_ * H_) / 256, 256, 0, stream>>>(Wg, 2 * H_, Wghi, Wglo, H_, H_);

    // GEMM1: text_proj = text @ Wt^T + bt  -> bf16 hi/lo
    gemm_kernel<<<dim3(H_ / 64, R_ / 256), 256, 0, stream>>>(
        Ahi, Alo, Wthi, Wtlo, 0, bt, TPhi, TPlo,
        nullptr, nullptr, nullptr, nullptr, nullptr);

    agree_kernel  <<<R_ / 8, 256, 0, stream>>>(TPhi, TPlo, vproj, agree);
    softmax_kernel<<<B_,     256, 0, stream>>>(agree, rw);

    // GEMM2: gate path + fused -> d_out
    gemm_kernel<<<dim3(H_ / 64, R_ / 256), 256, 0, stream>>>(
        TPhi, TPlo, Wghi, Wglo, 1, bg, nullptr, nullptr,
        rw, vpg, vproj, text, out);

    ln_kernel<<<R_, 256, 0, stream>>>(out, gamma, beta);
}
